// CNet_Conv2D_RCAB_STB_80118319939942
// MI455X (gfx1250) — compile-verified, matrix operand order checked by reading, not run on hardware
//
#include <hip/hip_runtime.h>
#include <hip/hip_bf16.h>

// ---------------------------------------------------------------------------
// CDNA5 (gfx1250): GEMM-like math through v_wmma_f32_16x16x32_bf16 (fp32 acc),
// weight tiles staged LDS-side via the Tensor Data Mover (tensor_load_to_lds,
// TENSORcnt), fragments built with ds_load_b128 per the 16-bit WMMA layout.
// ---------------------------------------------------------------------------

typedef __attribute__((ext_vector_type(16))) __bf16 v16bf;
typedef __attribute__((ext_vector_type(8)))  __bf16 v8bf;
typedef __attribute__((ext_vector_type(8)))  float  v8f;
typedef __attribute__((ext_vector_type(4)))  unsigned int u32x4;
typedef __attribute__((ext_vector_type(4)))  int i32x4;
typedef __attribute__((ext_vector_type(8)))  int i32x8;

#ifndef __has_builtin
#define __has_builtin(x) 0
#endif
#if __has_builtin(__builtin_amdgcn_tensor_load_to_lds)
#define HAVE_TDM 1
#else
#define HAVE_TDM 0
#endif

#define DEVI __device__ __forceinline__

DEVI __bf16 f2bf(float f) {
  unsigned u = __builtin_bit_cast(unsigned, f);
  unsigned r = u + 0x7FFFu + ((u >> 16) & 1u);   // round-to-nearest-even
  unsigned short h = (unsigned short)(r >> 16);
  return __builtin_bit_cast(__bf16, h);
}

DEVI void wait_tensorcnt0() {
#if __has_builtin(__builtin_amdgcn_s_wait_tensorcnt)
  __builtin_amdgcn_s_wait_tensorcnt(0);
#else
  asm volatile("s_wait_tensorcnt 0x0" ::: "memory");
#endif
}

#if HAVE_TDM
// TDM: DMA a 2D bf16 tile [tile_n rows x tile_k elems] (row stride
// row_stride_elems) from global into LDS at byte offset lds_off.
// D# bitfields per CDNA5 ISA ch.8 (count=1, data_size=2B, type=2).
DEVI void tdm_load_2d_bf16(unsigned lds_off, const void* gaddr,
                           unsigned tile_k, unsigned tile_n,
                           unsigned tensor_k, unsigned tensor_n,
                           unsigned row_stride_elems) {
  unsigned long long ga = (unsigned long long)(size_t)gaddr;
  u32x4 g0 = {};
  g0.x = 1u;                                          // count = 1 (valid D#)
  g0.y = lds_off;                                     // LDS byte address
  g0.z = (unsigned)(ga & 0xFFFFFFFFull);              // global_addr[31:0]
  g0.w = (unsigned)((ga >> 32) & 0x01FFFFFFull) | (2u << 30);  // [56:32]|type=2
  i32x8 g1 = {};
  g1[0] = (1 << 16);                                  // data_size = 1 -> 2 bytes
  g1[1] = (int)((tensor_k & 0xFFFFu) << 16);          // tensor_dim0[15:0]
  g1[2] = (int)((tensor_k >> 16) | ((tensor_n & 0xFFFFu) << 16));
  g1[3] = (int)((tensor_n >> 16) | (tile_k << 16));   // tile_dim0
  g1[4] = (int)(tile_n & 0xFFFFu);                    // tile_dim1 (tile_dim2=0)
  g1[5] = (int)row_stride_elems;                      // tensor_dim0_stride[31:0]
  g1[6] = 0;
  g1[7] = 0;
  i32x4 gz = {0, 0, 0, 0};
#if __clang_major__ >= 23
  i32x8 gz8 = {0, 0, 0, 0, 0, 0, 0, 0};
  __builtin_amdgcn_tensor_load_to_lds(g0, g1, gz, gz, gz8, 0);
#else
  __builtin_amdgcn_tensor_load_to_lds(g0, g1, gz, gz, 0);
#endif
}
#endif

// Build a 16x32 bf16 A/B fragment from an LDS row (K-contiguous).
// Lanes 0-15 hold K={0..7,16..23}, lanes 16-31 hold K={8..15,24..31}
// (ISA 7.12.2) -> two contiguous ds_load_b128 per lane.
DEVI v16bf frag_ld(const __bf16* row, int laneHi) {
  const v8bf* p = (const v8bf*)row;
  v8bf lo = p[laneHi];
  v8bf hi = p[2 + laneHi];
  return __builtin_shufflevector(lo, hi, 0, 1, 2, 3, 4, 5, 6, 7,
                                         8, 9, 10, 11, 12, 13, 14, 15);
}

DEVI v8f wmma_bf16(v16bf a, v16bf b, v8f c) {
  return __builtin_amdgcn_wmma_f32_16x16x32_bf16(
      false, a, false, b, (short)0, c, false, false);
}

enum { ACT_NONE = 0, ACT_RELU = 1, ACT_LRELU = 2, ACT_GELU = 3 };

template <int ACT> DEVI float act(float x) {
  if (ACT == ACT_RELU)  return x > 0.f ? x : 0.f;
  if (ACT == ACT_LRELU) return x > 0.f ? x : 0.2f * x;
  if (ACT == ACT_GELU)  return 0.5f * x * (1.f + erff(x * 0.70710678118f));
  return x;
}

// ---------------------------------------------------------------------------
// Weight prep (once per launch, tiny): fp32 [K,N] -> bf16 [N,K]; conv weights
// fp32 [Cout][Cin][KS*KS] -> bf16 [tap][Cout][Cin] (TDM-friendly 2D tiles).
// ---------------------------------------------------------------------------
__global__ void transpose_bf16_kernel(const float* __restrict__ B,
                                      __bf16* __restrict__ Bt, int K, int N) {
  long i = (long)blockIdx.x * blockDim.x + threadIdx.x;
  if (i >= (long)K * N) return;
  int k = (int)(i % K), n = (int)(i / K);
  Bt[i] = f2bf(B[(long)k * N + n]);
}

__global__ void conv_w_repack_kernel(const float* __restrict__ Ws,
                                     __bf16* __restrict__ Wd,
                                     int Cout, int Cin, int KS2) {
  long i = (long)blockIdx.x * blockDim.x + threadIdx.x;
  if (i >= (long)KS2 * Cout * Cin) return;
  int cc = (int)(i % Cin); long t = i / Cin;
  int m = (int)(t % Cout); int tap = (int)(t / Cout);
  Wd[i] = f2bf(Ws[((long)m * Cin + cc) * KS2 + tap]);
}

// ---------------------------------------------------------------------------
// Generic GEMM:  D[M,N] = act(A[M,K] @ B[K,N] + bias[N] (+ res[M,N]))
// A: fp32 (convert on stage).  Bt: bf16 [N,K] -> each K-tile is a contiguous
// 2D tile, DMA'd to LDS by the Tensor Data Mover (wave 0 issues, TENSORcnt).
// ---------------------------------------------------------------------------
template <int ACT>
__global__ __launch_bounds__(256) void gemm_wmma_kernel(
    const float* __restrict__ A, const __bf16* __restrict__ Bt,
    const float* __restrict__ bias, const float* __restrict__ res,
    float* __restrict__ D, int M, int N, int K) {
  __shared__ __bf16 sA[64 * 32];
  __shared__ __bf16 sB[64 * 32];  // B^T tile: [n][k]
  const int tid = threadIdx.x, lane = tid & 31, wid = tid >> 5;
  const int laneHi = (lane >> 4) & 1, ln = lane & 15;
  const int m0 = blockIdx.y * 64, n0 = blockIdx.x * 64;
  const int my = wid & 3, nx2 = wid >> 2;
  v8f acc0 = {}, acc1 = {};
  const int row = tid >> 2, c0 = (tid & 3) * 8;
  const __bf16 ZB = f2bf(0.f);
  for (int k0 = 0; k0 < K; k0 += 32) {
#if HAVE_TDM
    if (wid == 0)
      tdm_load_2d_bf16((unsigned)(size_t)sB, Bt + (long)n0 * K + k0,
                       32u, 64u, (unsigned)(K - k0), (unsigned)(N - n0),
                       (unsigned)K);
#endif
    // stage A tile (fp32 -> bf16)
    const int gm = m0 + row;
    if (k0 + 32 < K) __builtin_prefetch(&A[(long)gm * K + k0 + 32], 0, 1);
#pragma unroll
    for (int j = 0; j < 8; ++j) {
      int gk = k0 + c0 + j;
      float v = (gm < M && gk < K) ? A[(long)gm * K + gk] : 0.f;
      sA[row * 32 + c0 + j] = f2bf(v);
    }
#if !HAVE_TDM
    const int gn = n0 + row;
#pragma unroll
    for (int j = 0; j < 8; ++j) {
      int gk = k0 + c0 + j;
      sB[row * 32 + c0 + j] =
          (gn < N && gk < K) ? Bt[(long)gn * K + gk] : ZB;
    }
#else
    if (wid == 0) wait_tensorcnt0();
#endif
    __syncthreads();
    v16bf a  = frag_ld(&sA[(my * 16 + ln) * 32], laneHi);
    v16bf b0 = frag_ld(&sB[(nx2 * 32 + ln) * 32], laneHi);
    v16bf b1 = frag_ld(&sB[(nx2 * 32 + 16 + ln) * 32], laneHi);
    acc0 = wmma_bf16(a, b0, acc0);
    acc1 = wmma_bf16(a, b1, acc1);
    __syncthreads();
  }
#pragma unroll
  for (int r = 0; r < 8; ++r) {
    int rr = m0 + my * 16 + r + 8 * laneHi;
    if (rr >= M) continue;
    int ca = n0 + nx2 * 32 + ln;
    if (ca < N) {
      float v = acc0[r] + (bias ? bias[ca] : 0.f);
      if (res) v += res[(long)rr * N + ca];
      D[(long)rr * N + ca] = act<ACT>(v);
    }
    int cb = ca + 16;
    if (cb < N) {
      float v = acc1[r] + (bias ? bias[cb] : 0.f);
      if (res) v += res[(long)rr * N + cb];
      D[(long)rr * N + cb] = act<ACT>(v);
    }
  }
}

// ---------------------------------------------------------------------------
// Implicit-GEMM conv (NCHW, stride 1, pad KS/2).  M = Cout (<=96), N = 64
// pixels along one row, K = Cin per tap.  Repacked bf16 weights [tap][m][cc]
// are DMA'd per (tap, k-tile) by the TDM; input patch staged with coalesced
// fp32 loads (converted).  Coalesced NCHW stores.
// ---------------------------------------------------------------------------
template <int KS, int ACT>
__global__ __launch_bounds__(256) void conv_wmma_kernel(
    const float* __restrict__ X, const __bf16* __restrict__ Wb,
    const float* __restrict__ bias, const float* __restrict__ res,
    float* __restrict__ Y, int Bn, int Cin, int Cout, int H, int W) {
  __shared__ __bf16 sW[96 * 32];
  __shared__ __bf16 sX[64 * 32];  // [px][k]
  const int PAD = KS / 2;
  const int tid = threadIdx.x, lane = tid & 31, wid = tid >> 5;
  const int laneHi = (lane >> 4) & 1, ln = lane & 15;
  long blk = blockIdx.x;
  const int xb = (int)(blk % (W / 64)) * 64;
  const int y  = (int)((blk / (W / 64)) % H);
  const int b  = (int)(blk / ((long)(W / 64) * H));
  v8f acc[3] = {{}, {}, {}};
  int myA[3], nxA[3];
#pragma unroll
  for (int i = 0; i < 3; ++i) { int t = wid + 8 * i; myA[i] = t >> 2; nxA[i] = t & 3; }
  const int kTiles = (Cin + 31) >> 5;
  const __bf16 ZB = f2bf(0.f);
  for (int tap = 0; tap < KS * KS; ++tap) {
    const int ty = tap / KS - PAD, tx = tap % KS - PAD;
    const int yy = y + ty;
    for (int kt = 0; kt < kTiles; ++kt) {
      const int kc0 = kt << 5;
#if HAVE_TDM
      if (wid == 0)
        tdm_load_2d_bf16((unsigned)(size_t)sW,
                         Wb + (long)tap * Cout * Cin + kc0,
                         32u, 96u, (unsigned)(Cin - kc0), (unsigned)Cout,
                         (unsigned)Cin);
#else
#pragma unroll
      for (int i = 0; i < 12; ++i) {  // 96x32 weights
        int idx = tid + 256 * i;
        int m = idx >> 5, k = idx & 31, cc = kc0 + k;
        sW[idx] = (m < Cout && cc < Cin)
                      ? Wb[((long)tap * Cout + m) * Cin + cc] : ZB;
      }
#endif
#pragma unroll
      for (int j = 0; j < 8; ++j) {   // 64x32 input patch (coalesced on px)
        int idx = tid + 256 * j;
        int k = idx >> 6, px = idx & 63, cc = kc0 + k;
        int xx = xb + px + tx;
        float v = 0.f;
        if (cc < Cin && yy >= 0 && yy < H && xx >= 0 && xx < W)
          v = X[((long)(b * Cin + cc) * H + yy) * W + xx];
        sX[px * 32 + k] = f2bf(v);
      }
#if HAVE_TDM
      if (wid == 0) wait_tensorcnt0();
#endif
      __syncthreads();
#pragma unroll
      for (int i = 0; i < 3; ++i) {
        v16bf a  = frag_ld(&sW[(myA[i] * 16 + ln) * 32], laneHi);
        v16bf bb = frag_ld(&sX[(nxA[i] * 16 + ln) * 32], laneHi);
        acc[i] = wmma_bf16(a, bb, acc[i]);
      }
      __syncthreads();
    }
  }
#pragma unroll
  for (int i = 0; i < 3; ++i) {
#pragma unroll
    for (int r = 0; r < 8; ++r) {
      int oc = myA[i] * 16 + r + 8 * laneHi;
      int px = xb + nxA[i] * 16 + ln;
      if (oc < Cout) {
        long o = ((long)(b * Cout + oc) * H + y) * W + px;
        float v = acc[i][r] + (bias ? bias[oc] : 0.f);
        if (res) v += res[o];
        Y[o] = act<ACT>(v);
      }
    }
  }
}

// ---------------------------------------------------------------------------
// Window attention: one block per (head, window, batch).  QK^T + rel bias
// (staged in LDS) + shift mask + softmax + P·V, all via WMMA.
// ---------------------------------------------------------------------------
__global__ __launch_bounds__(128) void wmsa_attn_kernel(
    const float* __restrict__ qkv, const float* __restrict__ rel,
    float* __restrict__ out, int shifted) {
  __shared__ __bf16 sQ[64 * 32];
  __shared__ __bf16 sK[64 * 32];
  __shared__ __bf16 sVT[32 * 64];
  __shared__ float  sS[64 * 64];
  __shared__ __bf16 sP[64 * 64];
  __shared__ float  sRel[225];
  const int tid = threadIdx.x, lane = tid & 31, wid = tid >> 5;
  const int laneHi = (lane >> 4) & 1, ln = lane & 15;
  const int h = blockIdx.x, w = blockIdx.y, b = blockIdx.z;
  const long base = ((long)b * 1024 + w) * 64 * 243;
  for (int i = tid; i < 225; i += 128) sRel[i] = rel[h * 225 + i];
  for (int i = tid; i < 64 * 32; i += 128) {
    int p = i >> 5, c = i & 31;
    float q = (c < 27) ? qkv[base + p * 243 + h * 27 + c] : 0.f;
    float k = (c < 27) ? qkv[base + p * 243 + (3 + h) * 27 + c] : 0.f;
    sQ[i] = f2bf(q);
    sK[i] = f2bf(k);
  }
  for (int i = tid; i < 32 * 64; i += 128) {
    int c = i >> 6, p = i & 63;
    float v = (c < 27) ? qkv[base + p * 243 + (6 + h) * 27 + c] : 0.f;
    sVT[i] = f2bf(v);
  }
  __syncthreads();
  const float scl = 0.19245008973f;  // 27^-0.5
  const int wy = w >> 5, wx = w & 31;
  const int my = wid;
  for (int nx = 0; nx < 4; ++nx) {
    v16bf a  = frag_ld(&sQ[(my * 16 + ln) * 32], laneHi);
    v16bf bb = frag_ld(&sK[(nx * 16 + ln) * 32], laneHi);
    v8f acc = {};
    acc = wmma_bf16(a, bb, acc);
#pragma unroll
    for (int r = 0; r < 8; ++r) {
      int p = my * 16 + r + 8 * laneHi;
      int q = nx * 16 + ln;
      int py = p >> 3, px = p & 7, qy = q >> 3, qx = q & 7;
      float s = acc[r] * scl + sRel[(py - qy + 7) * 15 + (px - qx + 7)];
      if (shifted) {
        bool m = (wy == 31 && ((py < 4) != (qy < 4))) ||
                 (wx == 31 && ((px < 4) != (qx < 4)));
        if (m) s = -1e30f;
      }
      sS[p * 64 + q] = s;
    }
  }
  __syncthreads();
  if (tid < 64) {
    float mx = -1e30f;
    for (int q = 0; q < 64; ++q) mx = fmaxf(mx, sS[tid * 64 + q]);
    float sum = 0.f;
    for (int q = 0; q < 64; ++q) sum += expf(sS[tid * 64 + q] - mx);
    float inv = 1.f / sum;
    for (int q = 0; q < 64; ++q)
      sP[tid * 64 + q] = f2bf(expf(sS[tid * 64 + q] - mx) * inv);
  }
  __syncthreads();
  const long ob = ((long)b * 1024 + w) * 64;
  for (int nt = 0; nt < 2; ++nt) {
    v8f acc = {};
    for (int kt = 0; kt < 2; ++kt) {
      v16bf a  = frag_ld(&sP[(my * 16 + ln) * 64 + kt * 32], laneHi);
      v16bf bb = frag_ld(&sVT[(nt * 16 + ln) * 64 + kt * 32], laneHi);
      acc = wmma_bf16(a, bb, acc);
    }
#pragma unroll
    for (int r = 0; r < 8; ++r) {
      int p = my * 16 + r + 8 * laneHi;
      int c = nt * 16 + ln;
      if (c < 27) out[(ob + p) * 81 + h * 27 + c] = acc[r];
    }
  }
}

// ---------------------------------------------------------------------------
// LayerNorm over last dim (C=81), one wave32 per row.
// ---------------------------------------------------------------------------
__global__ __launch_bounds__(256) void layernorm_kernel(
    const float* __restrict__ X, const float* __restrict__ g,
    const float* __restrict__ bta, float* __restrict__ Y, long rows, int C) {
  long row = (long)blockIdx.x * 8 + (threadIdx.x >> 5);
  int lane = threadIdx.x & 31;
  if (row >= rows) return;
  const float* x = X + row * C;
  float s = 0.f, s2 = 0.f;
  for (int c = lane; c < C; c += 32) { float v = x[c]; s += v; s2 += v * v; }
  for (int o = 16; o > 0; o >>= 1) {
    s += __shfl_xor(s, o);
    s2 += __shfl_xor(s2, o);
  }
  float mu = s / C;
  float inv = rsqrtf(s2 / C - mu * mu + 1e-5f);
  float* y = Y + row * C;
  for (int c = lane; c < C; c += 32) y[c] = (x[c] - mu) * inv * g[c] + bta[c];
}

// ---------------------------------------------------------------------------
// Windowing (cyclic shift folded in), permutes, elementwise.
// ---------------------------------------------------------------------------
__global__ void window_gather_kernel(const float* __restrict__ X,
                                     float* __restrict__ XW, int Bn, int shift) {
  long i = (long)blockIdx.x * blockDim.x + threadIdx.x;
  long total = (long)Bn * 1024 * 64 * 81;
  if (i >= total) return;
  int c = (int)(i % 81); long t = i / 81;
  int p = (int)(t % 64); t /= 64;
  int w = (int)(t % 1024); t /= 1024;
  int b = (int)t;
  int yy = (((w >> 5) << 3) + (p >> 3) + shift) & 255;
  int xx = (((w & 31) << 3) + (p & 7) + shift) & 255;
  XW[i] = X[(((long)b * 256 + yy) * 256 + xx) * 81 + c];
}

__global__ void window_scatter_add_kernel(const float* __restrict__ P,
                                          const float* __restrict__ Xres,
                                          float* __restrict__ Y, int Bn, int shift) {
  long i = (long)blockIdx.x * blockDim.x + threadIdx.x;
  long total = (long)Bn * 256 * 256 * 81;
  if (i >= total) return;
  int c = (int)(i % 81); long t = i / 81;
  int x = (int)(t % 256); t /= 256;
  int y = (int)(t % 256); t /= 256;
  int b = (int)t;
  int yr = (y - shift) & 255, xr = (x - shift) & 255;
  int w = ((yr >> 3) << 5) + (xr >> 3);
  int p = ((yr & 7) << 3) + (xr & 7);
  Y[i] = Xres[i] + P[(((long)b * 1024 + w) * 64 + p) * 81 + c];
}

__global__ void nchw_to_nhwc_kernel(const float* __restrict__ X,
                                    float* __restrict__ Y, int Bn, int C, int H, int W) {
  long i = (long)blockIdx.x * blockDim.x + threadIdx.x;
  long total = (long)Bn * C * H * W;
  if (i >= total) return;
  int c = (int)(i % C); long t = i / C;
  int x = (int)(t % W); t /= W;
  int y = (int)(t % H); t /= H;
  int b = (int)t;
  Y[i] = X[((long)(b * C + c) * H + y) * W + x];
}

__global__ void nhwc_to_nchw_kernel(const float* __restrict__ X,
                                    float* __restrict__ Y, int Bn, int C, int H, int W) {
  long i = (long)blockIdx.x * blockDim.x + threadIdx.x;
  long total = (long)Bn * C * H * W;
  if (i >= total) return;
  int x = (int)(i % W); long t = i / W;
  int y = (int)(t % H); t /= H;
  int c = (int)(t % C); t /= C;
  int b = (int)t;
  Y[i] = X[(((long)b * H + y) * W + x) * C + c];
}

__global__ void scale_res_kernel(const float* __restrict__ X,
                                 const float* __restrict__ scale,
                                 const float* __restrict__ R,
                                 float* __restrict__ Y, int Bn, int C, long HW) {
  long i = (long)blockIdx.x * blockDim.x + threadIdx.x;
  long total = (long)Bn * C * HW;
  if (i >= total) return;
  int c = (int)((i / HW) % C);
  int b = (int)(i / (HW * C));
  Y[i] = X[i] * scale[b * C + c] + R[i];
}

__global__ void add_kernel(const float* __restrict__ A, const float* __restrict__ B,
                           float* __restrict__ Y, long n) {
  long i = (long)blockIdx.x * blockDim.x + threadIdx.x;
  if (i < n) Y[i] = A[i] + B[i];
}

__global__ void concat_c_kernel(const float* __restrict__ A, const float* __restrict__ B,
                                float* __restrict__ Y, int Bn, int C1, int C2, long HW) {
  long i = (long)blockIdx.x * blockDim.x + threadIdx.x;
  long total = (long)Bn * (C1 + C2) * HW;
  if (i >= total) return;
  long sp = i % HW;
  int c = (int)((i / HW) % (C1 + C2));
  int b = (int)(i / (HW * (C1 + C2)));
  Y[i] = (c < C1) ? A[((long)b * C1 + c) * HW + sp]
                  : B[((long)b * C2 + (c - C1)) * HW + sp];
}

// ---------------------------------------------------------------------------
// 256-point shared-memory radix-2 FFT (rows + columns).
// ---------------------------------------------------------------------------
DEVI int bitrev8(int x) {
  x = ((x & 0x0F) << 4) | ((x >> 4) & 0x0F);
  x = ((x & 0x33) << 2) | ((x >> 2) & 0x33);
  x = ((x & 0x55) << 1) | ((x >> 1) & 0x55);
  return x;
}

template <int DIR>
DEVI void fft256(float* re, float* im, int tid) {
  for (int i = tid; i < 256; i += 128) {
    int j = bitrev8(i);
    if (j > i) {
      float tr = re[i]; re[i] = re[j]; re[j] = tr;
      float ti = im[i]; im[i] = im[j]; im[j] = ti;
    }
  }
  __syncthreads();
  for (int len = 2; len <= 256; len <<= 1) {
    int half = len >> 1;
    int grp = tid / half, pos = tid % half;
    int i0 = grp * len + pos, i1 = i0 + half;
    float ang = (DIR > 0 ? -2.f : 2.f) * 3.14159265358979f * pos / len;
    float wr = cosf(ang), wi = sinf(ang);
    float xr = re[i1], xi = im[i1];
    float tr = wr * xr - wi * xi, ti = wr * xi + wi * xr;
    float ur = re[i0], ui = im[i0];
    re[i0] = ur + tr; im[i0] = ui + ti;
    re[i1] = ur - tr; im[i1] = ui - ti;
    __syncthreads();
  }
}

__global__ __launch_bounds__(128) void fft_rows_fwd_kernel(
    const float* __restrict__ X, float* __restrict__ F, int Bn, int Cm) {
  __shared__ float re[256], im[256];
  long blk = blockIdx.x;
  int y = (int)(blk % 256); long t = blk / 256;
  int c = (int)(t % Cm); int b = (int)(t / Cm);
  const float* x = X + ((long)(b * Cm + c) * 256 + y) * 256;
  for (int i = threadIdx.x; i < 256; i += 128) { re[i] = x[i] * 0.0625f; im[i] = 0.f; }
  __syncthreads();
  fft256<1>(re, im, threadIdx.x);
  __syncthreads();
  float* fr = F + (((long)(b * Cm + c) * 2 + 0) * 256 + y) * 256;
  float* fi = F + (((long)(b * Cm + c) * 2 + 1) * 256 + y) * 256;
  for (int i = threadIdx.x; i < 256; i += 128) { fr[i] = re[i]; fi[i] = im[i]; }
}

template <int DIR>
__global__ __launch_bounds__(128) void fft_cols_kernel(
    const float* __restrict__ Fin, float* __restrict__ Fout, int Bn, int Cm) {
  __shared__ float re[256], im[256];
  long blk = blockIdx.x;
  int x = (int)(blk % 256); long t = blk / 256;
  int c = (int)(t % Cm); int b = (int)(t / Cm);
  const float* fr = Fin + (((long)(b * Cm + c) * 2 + 0) * 256) * 256 + x;
  const float* fi = Fin + (((long)(b * Cm + c) * 2 + 1) * 256) * 256 + x;
  for (int i = threadIdx.x; i < 256; i += 128) {
    re[i] = fr[(long)i * 256] * 0.0625f;
    im[i] = fi[(long)i * 256] * 0.0625f;
  }
  __syncthreads();
  fft256<DIR>(re, im, threadIdx.x);
  __syncthreads();
  float* gr = Fout + (((long)(b * Cm + c) * 2 + 0) * 256) * 256 + x;
  float* gi = Fout + (((long)(b * Cm + c) * 2 + 1) * 256) * 256 + x;
  for (int i = threadIdx.x; i < 256; i += 128) {
    gr[(long)i * 256] = re[i];
    gi[(long)i * 256] = im[i];
  }
}

__global__ __launch_bounds__(128) void fft_rows_inv_real_kernel(
    const float* __restrict__ F, float* __restrict__ Y, int Bn, int Cm) {
  __shared__ float re[256], im[256];
  long blk = blockIdx.x;
  int y = (int)(blk % 256); long t = blk / 256;
  int c = (int)(t % Cm); int b = (int)(t / Cm);
  const float* fr = F + (((long)(b * Cm + c) * 2 + 0) * 256 + y) * 256;
  const float* fi = F + (((long)(b * Cm + c) * 2 + 1) * 256 + y) * 256;
  for (int i = threadIdx.x; i < 256; i += 128) {
    re[i] = fr[i] * 0.0625f;
    im[i] = fi[i] * 0.0625f;
  }
  __syncthreads();
  fft256<-1>(re, im, threadIdx.x);
  __syncthreads();
  float* yo = Y + ((long)(b * Cm + c) * 256 + y) * 256;
  for (int i = threadIdx.x; i < 256; i += 128) yo[i] = re[i];
}

// ---------------------------------------------------------------------------
// Direct conv (prompt extractor) + spatial mean.
// ---------------------------------------------------------------------------
template <int ACT>
__global__ void conv_direct_kernel(const float* __restrict__ X,
                                   const float* __restrict__ Wt,
                                   const float* __restrict__ bias,
                                   float* __restrict__ Y, int Bn, int Cin, int Cout,
                                   int Hin, int Win, int KS, int S, int PAD,
                                   int Hout, int Wout) {
  long i = (long)blockIdx.x * blockDim.x + threadIdx.x;
  long total = (long)Bn * Cout * Hout * Wout;
  if (i >= total) return;
  int x = (int)(i % Wout); long t = i / Wout;
  int y = (int)(t % Hout); t /= Hout;
  int oc = (int)(t % Cout); int b = (int)(t / Cout);
  float s = bias ? bias[oc] : 0.f;
  for (int ic = 0; ic < Cin; ++ic)
    for (int ky = 0; ky < KS; ++ky) {
      int yy = y * S + ky - PAD;
      if (yy < 0 || yy >= Hin) continue;
      for (int kx = 0; kx < KS; ++kx) {
        int xx = x * S + kx - PAD;
        if (xx < 0 || xx >= Win) continue;
        s += X[((long)(b * Cin + ic) * Hin + yy) * Win + xx] *
             Wt[(((long)oc * Cin + ic) * KS + ky) * KS + kx];
      }
    }
  Y[i] = act<ACT>(s);
}

__global__ __launch_bounds__(256) void mean_hw_kernel(const float* __restrict__ X,
                                                      float* __restrict__ Y, int HW) {
  int bc = blockIdx.x;
  const float* x = X + (long)bc * HW;
  float s = 0.f;
  for (int i = threadIdx.x; i < HW; i += 256) s += x[i];
  __shared__ float red[256];
  red[threadIdx.x] = s;
  __syncthreads();
  for (int o = 128; o > 0; o >>= 1) {
    if (threadIdx.x < (unsigned)o) red[threadIdx.x] += red[threadIdx.x + o];
    __syncthreads();
  }
  if (threadIdx.x == 0) Y[bc] = red[0] / HW;
}

// ---------------------------------------------------------------------------
// Orchestration
// ---------------------------------------------------------------------------
static inline dim3 ew_grid(long n) { return dim3((unsigned)((n + 255) / 256)); }

extern "C" void kernel_launch(void* const* d_in, const int* in_sizes, int n_in,
                              void* d_out, int out_size, void* d_ws, size_t ws_size,
                              hipStream_t stream) {
  const int Bn = 2, C = 81, H = 256, W = 256, Cm = 40;
  const long HW = (long)H * W;
  const long FMAP = (long)Bn * C * HW;    // 10,616,832
  const long ROWS = (long)Bn * HW;        // 131072 token rows

  const float* Wx    = (const float*)d_in[0];
  const float* angle = (const float*)d_in[1];
  auto P = [&](int i) { return (const float*)d_in[i]; };
  const int LOW_W1 = 2, LOW_B1 = 3, LOW_W2 = 4, LOW_B2 = 5;
  const int STB0 = 6, STB1 = 19;
  const int RES_W1 = 32, RES_B1 = 33, RES_W2 = 34, RES_B2 = 35;
  const int ST_W1 = 36, ST_B1 = 37, FU_W = 38, FU_B = 39, ST_W2 = 40, ST_B2 = 41;
  const int FUS_W = 42, FUS_B = 43;
  const int FE_W1 = 44, FE_B1 = 45, FE_W2 = 46, FE_B2 = 47, FE_W3 = 48, FE_B3 = 49;
  const int PS_W = 50, PS_B = 51, LAST_W = 52, LAST_B = 53;

  // workspace layout (floats)
  float* w = (float*)d_ws;
  float* t0 = w;
  float* t1 = t0 + FMAP;
  float* t2 = t1 + FMAP;
  float* t3 = t2 + FMAP;
  float* x2b = t3 + FMAP;
  float* G = x2b + FMAP;                 // 131072*324 GEMM scratch
  float* freqA = G + ROWS * 324;
  float* freqB = freqA + (long)Bn * 2 * Cm * HW;
  float* f1b = freqB + (long)Bn * 2 * Cm * HW;
  float* fub = f1b + (long)Bn * Cm * HW;
  float* sb = fub + (long)Bn * Cm * HW;
  float* fb = sb + FMAP;
  float* catb = fb + FMAP;
  float* feA = catb + (long)Bn * 2 * C * HW;
  float* feB = feA + (long)Bn * C * 126 * 126;
  float* feC = feB + (long)Bn * C * 63 * 63;
  float* prompt = feC + (long)Bn * C * 32 * 32;
  float* scale1 = prompt + 256;

  // bf16 weight arena
  __bf16* bfbase = (__bf16*)(scale1 + 256);
  long bfoff = 0;
  auto TW = [&](int idx, int K, int N) {  // fp32 [K,N] -> bf16 [N,K]
    __bf16* dst = bfbase + bfoff;
    bfoff = (bfoff + (long)K * N + 31) & ~31L;
    transpose_bf16_kernel<<<ew_grid((long)K * N), 256, 0, stream>>>(P(idx), dst, K, N);
    return dst;
  };
  auto CW = [&](int idx, int Cout, int Cin, int KS2) {  // -> bf16 [tap][m][cc]
    __bf16* dst = bfbase + bfoff;
    bfoff = (bfoff + (long)KS2 * Cout * Cin + 31) & ~31L;
    conv_w_repack_kernel<<<ew_grid((long)KS2 * Cout * Cin), 256, 0, stream>>>(
        P(idx), dst, Cout, Cin, KS2);
    return dst;
  };

  // ---- weight prep (tiny, once per launch) ------------------------------
  const __bf16* Wps = TW(PS_W, C, C);
  const __bf16* Wqkv[2] = {TW(STB0 + 2, C, 3 * C), TW(STB1 + 2, C, 3 * C)};
  const __bf16* Wproj[2] = {TW(STB0 + 5, C, C), TW(STB1 + 5, C, C)};
  const __bf16* Wm1[2] = {TW(STB0 + 9, C, 4 * C), TW(STB1 + 9, C, 4 * C)};
  const __bf16* Wm2[2] = {TW(STB0 + 11, 4 * C, C), TW(STB1 + 11, 4 * C, C)};
  const __bf16* Wlow1 = CW(LOW_W1, C, C, 9);
  const __bf16* Wlow2 = CW(LOW_W2, C, C, 9);
  const __bf16* Wres1 = CW(RES_W1, C, C, 9);
  const __bf16* Wres2 = CW(RES_W2, C, C, 9);
  const __bf16* Wst1 = CW(ST_W1, Cm, C, 1);
  const __bf16* Wfu = CW(FU_W, 2 * Cm, 2 * Cm, 1);
  const __bf16* Wst2 = CW(ST_W2, C, Cm, 1);
  const __bf16* Wfus = CW(FUS_W, C, 2 * C, 1);
  const __bf16* Wlast = CW(LAST_W, C, C, 9);

  float* x5 = (float*)d_out;
  float* prompt_out = x5 + FMAP;

  // ---- prompt extractor -------------------------------------------------
  conv_direct_kernel<ACT_RELU><<<ew_grid((long)Bn * C * 126 * 126), 256, 0, stream>>>(
      angle, P(FE_W1), P(FE_B1), feA, Bn, 1, C, 256, 256, 7, 2, 1, 126, 126);
  conv_direct_kernel<ACT_RELU><<<ew_grid((long)Bn * C * 63 * 63), 256, 0, stream>>>(
      feA, P(FE_W2), P(FE_B2), feB, Bn, C, C, 126, 126, 3, 2, 1, 63, 63);
  conv_direct_kernel<ACT_RELU><<<ew_grid((long)Bn * C * 32 * 32), 256, 0, stream>>>(
      feB, P(FE_W3), P(FE_B3), feC, Bn, C, C, 63, 63, 3, 2, 1, 32, 32);
  mean_hw_kernel<<<Bn * C, 256, 0, stream>>>(feC, prompt, 32 * 32);
  hipMemcpyAsync(prompt_out, prompt, (size_t)Bn * C * sizeof(float),
                 hipMemcpyDeviceToDevice, stream);
  gemm_wmma_kernel<ACT_NONE><<<dim3(2, 1), 256, 0, stream>>>(
      prompt, Wps, P(PS_B), nullptr, scale1, Bn, C, C);

  // ---- shallow convs + prompt modulation --------------------------------
  conv_wmma_kernel<3, ACT_RELU><<<Bn * H * (W / 64), 256, 0, stream>>>(
      Wx, Wlow1, P(LOW_B1), nullptr, t0, Bn, C, C, H, W);
  conv_wmma_kernel<3, ACT_NONE><<<Bn * H * (W / 64), 256, 0, stream>>>(
      t0, Wlow2, P(LOW_B2), nullptr, t1, Bn, C, C, H, W);
  scale_res_kernel<<<ew_grid(FMAP), 256, 0, stream>>>(t1, scale1, Wx, x2b, Bn, C, HW);
  nchw_to_nhwc_kernel<<<ew_grid(FMAP), 256, 0, stream>>>(x2b, t2, Bn, C, H, W);

  // ---- two Swin transformer blocks --------------------------------------
  for (int sbk = 0; sbk < 2; ++sbk) {
    const int base = (sbk == 0) ? STB0 : STB1;
    const int shifted = sbk;
    const int shift = shifted ? 4 : 0;
    layernorm_kernel<<<(unsigned)((ROWS + 7) / 8), 256, 0, stream>>>(
        t2, P(base + 0), P(base + 1), t3, ROWS, C);
    window_gather_kernel<<<ew_grid(ROWS * C), 256, 0, stream>>>(t3, t1, Bn, shift);
    gemm_wmma_kernel<ACT_NONE><<<dim3(4, (unsigned)(ROWS / 64)), 256, 0, stream>>>(
        t1, Wqkv[sbk], P(base + 3), nullptr, G, (int)ROWS, 3 * C, C);
    wmsa_attn_kernel<<<dim3(3, 1024, Bn), 128, 0, stream>>>(G, P(base + 4), t3, shifted);
    gemm_wmma_kernel<ACT_NONE><<<dim3(2, (unsigned)(ROWS / 64)), 256, 0, stream>>>(
        t3, Wproj[sbk], P(base + 6), nullptr, t1, (int)ROWS, C, C);
    window_scatter_add_kernel<<<ew_grid(ROWS * C), 256, 0, stream>>>(t1, t2, t0, Bn, shift);
    layernorm_kernel<<<(unsigned)((ROWS + 7) / 8), 256, 0, stream>>>(
        t0, P(base + 7), P(base + 8), t3, ROWS, C);
    gemm_wmma_kernel<ACT_GELU><<<dim3(6, (unsigned)(ROWS / 64)), 256, 0, stream>>>(
        t3, Wm1[sbk], P(base + 10), nullptr, G, (int)ROWS, 4 * C, C);
    gemm_wmma_kernel<ACT_NONE><<<dim3(2, (unsigned)(ROWS / 64)), 256, 0, stream>>>(
        G, Wm2[sbk], P(base + 12), t0, t2, (int)ROWS, C, 4 * C);
  }
  nhwc_to_nchw_kernel<<<ew_grid(FMAP), 256, 0, stream>>>(t2, t0, Bn, C, H, W);  // x3

  // ---- SFB: residual conv branch ----------------------------------------
  conv_wmma_kernel<3, ACT_LRELU><<<Bn * H * (W / 64), 256, 0, stream>>>(
      t0, Wres1, P(RES_B1), nullptr, t1, Bn, C, C, H, W);
  conv_wmma_kernel<3, ACT_NONE><<<Bn * H * (W / 64), 256, 0, stream>>>(
      t1, Wres2, P(RES_B2), t0, sb, Bn, C, C, H, W);

  // ---- SFB: Fourier branch ----------------------------------------------
  conv_wmma_kernel<1, ACT_LRELU><<<Bn * H * (W / 64), 256, 0, stream>>>(
      t0, Wst1, P(ST_B1), nullptr, f1b, Bn, C, Cm, H, W);
  fft_rows_fwd_kernel<<<(unsigned)(Bn * Cm * H), 128, 0, stream>>>(f1b, freqA, Bn, Cm);
  fft_cols_kernel<1><<<(unsigned)(Bn * Cm * W), 128, 0, stream>>>(freqA, freqB, Bn, Cm);
  conv_wmma_kernel<1, ACT_LRELU><<<Bn * H * (W / 64), 256, 0, stream>>>(
      freqB, Wfu, P(FU_B), nullptr, freqA, Bn, 2 * Cm, 2 * Cm, H, W);
  fft_cols_kernel<-1><<<(unsigned)(Bn * Cm * W), 128, 0, stream>>>(freqA, freqB, Bn, Cm);
  fft_rows_inv_real_kernel<<<(unsigned)(Bn * Cm * H), 128, 0, stream>>>(freqB, fub, Bn, Cm);
  add_kernel<<<ew_grid((long)Bn * Cm * HW), 256, 0, stream>>>(
      f1b, fub, fub, (long)Bn * Cm * HW);
  conv_wmma_kernel<1, ACT_NONE><<<Bn * H * (W / 64), 256, 0, stream>>>(
      fub, Wst2, P(ST_B2), nullptr, fb, Bn, Cm, C, H, W);

  // ---- fuse + last conv --------------------------------------------------
  concat_c_kernel<<<ew_grid((long)Bn * 2 * C * HW), 256, 0, stream>>>(
      sb, fb, catb, Bn, C, C, HW);
  conv_wmma_kernel<1, ACT_NONE><<<Bn * H * (W / 64), 256, 0, stream>>>(
      catb, Wfus, P(FUS_B), nullptr, t1, Bn, 2 * C, C, H, W);
  add_kernel<<<ew_grid(FMAP), 256, 0, stream>>>(t1, x2b, t0, FMAP);
  conv_wmma_kernel<3, ACT_NONE><<<Bn * H * (W / 64), 256, 0, stream>>>(
      t0, Wlast, P(LAST_B), nullptr, x5, Bn, C, C, H, W);
}
